// CasualMultiHeadAttention_28467043238064
// MI455X (gfx1250) — matrix operand order repaired and checked by reading, not compiled, not run on hardware
//
#include <hip/hip_runtime.h>
#include <hip/hip_bf16.h>

typedef __attribute__((ext_vector_type(16))) __bf16 v16bf;
typedef __attribute__((ext_vector_type(8)))  __bf16 v8bf;
typedef __attribute__((ext_vector_type(8)))  float  v8f;
typedef __attribute__((ext_vector_type(4)))  unsigned int v4u;
typedef __attribute__((ext_vector_type(8)))  int v8i;
typedef __attribute__((ext_vector_type(4)))  int v4i;

union bf16x16 { v16bf v; v8bf h[2]; };
union bf16x8  { v8bf v; unsigned short u[8]; };

__device__ __forceinline__ unsigned short f2bf(float f) {
  unsigned u = __float_as_uint(f);
  u += 0x7FFFu + ((u >> 16) & 1u);          // round-to-nearest-even
  return (unsigned short)(u >> 16);
}
__device__ __forceinline__ float bf2f(unsigned short h) {
  return __uint_as_float(((unsigned)h) << 16);
}

// LDS aperture: flat addr low 32 bits are the LDS byte address.
__device__ __forceinline__ unsigned lds_addr32(const void* p) {
  return (unsigned)(unsigned long long)p;
}

// Async global->LDS copy (CDNA5), tracked by ASYNCcnt.
__device__ __forceinline__ void async_copy_b128(unsigned lds, unsigned long long gaddr) {
  asm volatile("global_load_async_to_lds_b128 %0, %1, off"
               :: "v"(lds), "v"(gaddr) : "memory");
}
__device__ __forceinline__ void wait_async0() {
  asm volatile("s_wait_asynccnt 0x0" ::: "memory");
}

// XOR lane swap within 32-lane group via ds_swizzle immediate patterns.
template<int IMM>
__device__ __forceinline__ float swz(float x) {
  return __int_as_float(__builtin_amdgcn_ds_swizzle(__float_as_int(x), IMM));
}
#define SWAPX1 0x041F
#define SWAPX2 0x081F
#define SWAPX4 0x101F
#define SWAPX8 0x201F

// ---------------------------------------------------------------- convert
__global__ __launch_bounds__(256)
void f32_to_bf16_kernel(const float* __restrict__ in,
                        unsigned short* __restrict__ out, int n) {
  int i = blockIdx.x * blockDim.x + threadIdx.x;
  if (i < n) out[i] = f2bf(in[i]);
}

// ---------------------------------------------------------------- GEMM
// C[M=8192, N=1024] = A[M,1024](bf16) * W[N,1024]^T(bf16)
// Async-to-LDS double buffered K loop.
// MODE 0: store bf16 head-split [B,H,S,64];  MODE 1: store f32 row-major.
template<int MODE>
__global__ __launch_bounds__(256)
void wmma_gemm_kernel(const unsigned short* __restrict__ A,
                      const unsigned short* __restrict__ W,
                      void* __restrict__ Out)
{
  __shared__ __align__(16) unsigned short As[2][128 * 32];
  __shared__ __align__(16) unsigned short Bs[2][64 * 32];

  const int tid  = threadIdx.x;
  const int wv   = tid >> 5, lane = tid & 31;
  const int hl   = lane >> 4, l15 = lane & 15;
  const int m0   = blockIdx.x * 128, n0 = blockIdx.y * 64;
  const int mt   = (wv >> 1) * 32, nt = (wv & 1) * 32;

  const v8f vzero = {0.f,0.f,0.f,0.f,0.f,0.f,0.f,0.f};
  v8f acc[2][2];
#pragma unroll
  for (int i = 0; i < 2; ++i)
#pragma unroll
    for (int j = 0; j < 2; ++j) acc[i][j] = vzero;

  auto issue_tile = [&](int buf, int k0) {
#pragma unroll
    for (int c = 0; c < 2; ++c) {            // A tile: 128x32, 2 chunks/thread
      int chunk = tid + c * 256;
      int row = chunk >> 2, kc = (chunk & 3) << 3;
      async_copy_b128(lds_addr32(&As[buf][row * 32 + kc]),
                      (unsigned long long)&A[(size_t)(m0 + row) * 1024 + k0 + kc]);
    }
    int row = tid >> 2, kc = (tid & 3) << 3; // W tile: 64x32, 1 chunk/thread
    async_copy_b128(lds_addr32(&Bs[buf][row * 32 + kc]),
                    (unsigned long long)&W[(size_t)(n0 + row) * 1024 + k0 + kc]);
  };

  issue_tile(0, 0);
  wait_async0();
  __syncthreads();

  for (int kt = 0; kt < 32; ++kt) {
    const int cur = kt & 1;
    if (kt + 1 < 32) issue_tile(cur ^ 1, (kt + 1) * 32);

    const unsigned short* as = As[cur];
    const unsigned short* bs = Bs[cur];
    bf16x16 af[2], bw[2];
#pragma unroll
    for (int im = 0; im < 2; ++im) {     // A-operand: lane=row, K chunks of 8
      int row = mt + im * 16 + l15, kb = hl * 8;
      af[im].h[0] = *(const v8bf*)&as[row * 32 + kb];
      af[im].h[1] = *(const v8bf*)&as[row * 32 + kb + 16];
    }
#pragma unroll
    for (int in = 0; in < 2; ++in) {     // B-operand: lane=col, K chunks of 8
      int col = nt + in * 16 + l15, kb = hl * 8;
      bw[in].h[0] = *(const v8bf*)&bs[col * 32 + kb];
      bw[in].h[1] = *(const v8bf*)&bs[col * 32 + kb + 16];
    }
#pragma unroll
    for (int im = 0; im < 2; ++im)
#pragma unroll
      for (int in = 0; in < 2; ++in)
        acc[im][in] = __builtin_amdgcn_wmma_f32_16x16x32_bf16(
            false, af[im].v, false, bw[in].v, (short)0, acc[im][in],
            false, false);

    wait_async0();        // next tile resident before buffer swap
    __syncthreads();
  }

  // C layout: VGPR v -> row v + 8*hl, lane -> col l15
#pragma unroll
  for (int im = 0; im < 2; ++im)
#pragma unroll
    for (int in = 0; in < 2; ++in)
#pragma unroll
      for (int v = 0; v < 8; ++v) {
        int gm = m0 + mt + im * 16 + v + 8 * hl;
        int gn = n0 + nt + in * 16 + l15;
        float val = acc[im][in][v];
        if (MODE == 0) {
          int b = gm >> 11, s = gm & 2047, h = gn >> 6, d = gn & 63;
          ((unsigned short*)Out)[(((size_t)(b * 16 + h)) * 2048 + s) * 64 + d] =
              f2bf(val);
        } else {
          ((float*)Out)[(size_t)gm * 1024 + gn] = val;
        }
      }
}

// ---------------------------------------------------------------- RoPE
__global__ __launch_bounds__(256)
void rope_kernel(unsigned short* __restrict__ Q, unsigned short* __restrict__ K)
{
  int i = blockIdx.x * blockDim.x + threadIdx.x;   // < 4*16*2048*32
  int p = i & 31;
  int s = (i >> 5) & 2047;
  size_t base = (size_t)(i >> 5) * 64 + (p << 1);
  float ang = (float)s * __expf(-(float)p * 0.28782313662425572f);
  float c, sn;
  __sincosf(ang, &sn, &c);
  float qe = bf2f(Q[base]), qo = bf2f(Q[base + 1]);
  Q[base]     = f2bf(qe * c - qo * sn);
  Q[base + 1] = f2bf(qe * sn + qo * c);
  float ke = bf2f(K[base]), ko = bf2f(K[base + 1]);
  K[base]     = f2bf(ke * c - ko * sn);
  K[base + 1] = f2bf(ke * sn + ko * c);
}

// ---------------------------------------------------------------- flash attention
// grid (B*H, S/128), block 256. K tile fetched by the Tensor Data Mover
// (one descriptor per 32-key tile, hardware LDS padding to stride 72).
__global__ __launch_bounds__(256)
void flash_attn_kernel(const unsigned short* __restrict__ Qb,
                       const unsigned short* __restrict__ Kb,
                       const unsigned short* __restrict__ Vb,
                       unsigned short* __restrict__ Ob)
{
  constexpr int KSTR = 72;   // 64 els + 8-el pad -> 144B rows
  constexpr int VSTR = 40;   // V^T rows, 80B
  __shared__ __align__(16) unsigned short Kt[32 * KSTR];   // [key][d]
  __shared__ __align__(16) unsigned short Vt[64 * VSTR];   // [d][key]
  __shared__ __align__(16) unsigned short Ps[8][16 * 32];  // per-wave P

  const int bh = blockIdx.x;
  const int b  = bh >> 4, h = bh & 15;
  const int tid = threadIdx.x, wv = tid >> 5, lane = tid & 31;
  const int hl = lane >> 4, l15 = lane & 15;
  const int q0 = blockIdx.y * 128 + wv * 16;
  const size_t hbase = (size_t)bh * (2048 * 64);
  const float NEG_INF = -__builtin_inff();
  const unsigned kt_lds = lds_addr32(Kt);

  bf16x16 qf[2];
  {
    const v8bf* qp = (const v8bf*)(Qb + hbase + (size_t)(q0 + l15) * 64);
    qf[0].h[0] = qp[hl];      qf[0].h[1] = qp[hl + 2];
    qf[1].h[0] = qp[hl + 4];  qf[1].h[1] = qp[hl + 6];
  }

  const v8f vzero = {0.f,0.f,0.f,0.f,0.f,0.f,0.f,0.f};
  v8f o[4];
#pragma unroll
  for (int t = 0; t < 4; ++t) o[t] = vzero;
  float mrun[8], lrun[8];
#pragma unroll
  for (int v = 0; v < 8; ++v) { mrun[v] = NEG_INF; lrun[v] = 0.f; }

  const int qlast = q0 + 15;
  const int nkt = (blockIdx.y + 1) * 4;
  for (int kt = 0; kt < nkt; ++kt) {
    const int kv0 = kt * 32;

#if __has_builtin(__builtin_amdgcn_tensor_load_to_lds)
    if (wv == 0) {   // TDM: 32x64 bf16 tile of K, HW pad 4 dwords every 32
      unsigned long long ga = (unsigned long long)(Kb + hbase + (size_t)kv0 * 64);
      v4u g0 = { 1u, kt_lds, (unsigned)ga, (unsigned)(ga >> 32) | (2u << 30) };
      v8i g1 = { (int)0x07110000,        // data_size=2B, pad_en, intv=32dw, amt=4dw
                 (int)(64u << 16),       // tensor_dim0 = 64
                 (int)(2048u << 16),     // tensor_dim1 = 2048
                 (int)(64u << 16),       // tile_dim0 = 64
                 32,                     // tile_dim1 = 32
                 64, 0, 0 };             // tensor_dim0_stride = 64
      v4i gz  = { 0, 0, 0, 0 };
      v8i gz8 = { 0, 0, 0, 0, 0, 0, 0, 0 };
      __builtin_amdgcn_tensor_load_to_lds(g0, g1, gz, gz, gz8, 0);
    }
#else
    {
      int row = tid >> 3, ch = (tid & 7) << 3;
      *(v8bf*)&Kt[row * KSTR + ch] =
          *(const v8bf*)(Kb + hbase + (size_t)(kv0 + row) * 64 + ch);
    }
#endif
    {   // V tile loaded + transposed by all threads
      int row = tid >> 3, ch = (tid & 7) << 3;
      bf16x8 uv;
      uv.v = *(const v8bf*)(Vb + hbase + (size_t)(kv0 + row) * 64 + ch);
#pragma unroll
      for (int j = 0; j < 8; ++j) Vt[(ch + j) * VSTR + row] = uv.u[j];
    }
#if __has_builtin(__builtin_amdgcn_tensor_load_to_lds)
    if (wv == 0) __builtin_amdgcn_s_wait_tensorcnt(0);
#endif
    __syncthreads();

    if (kv0 <= qlast) {
      v8f sc0 = vzero, sc1 = vzero;
#pragma unroll
      for (int kn = 0; kn < 2; ++kn) {
        const unsigned short* kr = &Kt[(kn * 16 + l15) * KSTR];
        bf16x16 bA, bB;
        int d0 = hl * 8;
        bA.h[0] = *(const v8bf*)&kr[d0];
        bA.h[1] = *(const v8bf*)&kr[d0 + 16];
        bB.h[0] = *(const v8bf*)&kr[32 + d0];
        bB.h[1] = *(const v8bf*)&kr[32 + d0 + 16];
        v8f s = (kn == 0) ? sc0 : sc1;
        s = __builtin_amdgcn_wmma_f32_16x16x32_bf16(false, qf[0].v, false,
                                                    bA.v, (short)0, s, false, false);
        s = __builtin_amdgcn_wmma_f32_16x16x32_bf16(false, qf[1].v, false,
                                                    bB.v, (short)0, s, false, false);
        if (kn == 0) sc0 = s; else sc1 = s;
      }

      float p0[8], p1[8], corr[8];
#pragma unroll
      for (int v = 0; v < 8; ++v) {
        int row = q0 + v + 8 * hl;
        float s0 = (kv0 + l15      <= row) ? sc0[v] * 0.125f : NEG_INF;
        float s1 = (kv0 + 16 + l15 <= row) ? sc1[v] * 0.125f : NEG_INF;
        float t = fmaxf(s0, s1);
        t = fmaxf(t, swz<SWAPX1>(t));
        t = fmaxf(t, swz<SWAPX2>(t));
        t = fmaxf(t, swz<SWAPX4>(t));
        t = fmaxf(t, swz<SWAPX8>(t));
        float mn = fmaxf(mrun[v], t);
        corr[v] = __expf(mrun[v] - mn);
        mrun[v] = mn;
        float e0 = __expf(s0 - mn), e1 = __expf(s1 - mn);
        p0[v] = e0; p1[v] = e1;
        float su = e0 + e1;
        su += swz<SWAPX1>(su);
        su += swz<SWAPX2>(su);
        su += swz<SWAPX4>(su);
        su += swz<SWAPX8>(su);
        lrun[v] = lrun[v] * corr[v] + su;
      }
#pragma unroll
      for (int t = 0; t < 4; ++t)
#pragma unroll
        for (int v = 0; v < 8; ++v) o[t][v] = o[t][v] * corr[v];

      unsigned short* pw = &Ps[wv][0];
#pragma unroll
      for (int v = 0; v < 8; ++v) {
        int r = v + 8 * hl;
        pw[r * 32 + l15]      = f2bf(p0[v]);
        pw[r * 32 + 16 + l15] = f2bf(p1[v]);
      }
      bf16x16 pf;
      {
        int kb = hl * 8;
        pf.h[0] = *(const v8bf*)&pw[l15 * 32 + kb];
        pf.h[1] = *(const v8bf*)&pw[l15 * 32 + kb + 16];
      }
#pragma unroll
      for (int t = 0; t < 4; ++t) {
        const unsigned short* vr = &Vt[(t * 16 + l15) * VSTR];
        bf16x16 vf;
        int kk = hl * 8;
        vf.h[0] = *(const v8bf*)&vr[kk];
        vf.h[1] = *(const v8bf*)&vr[kk + 16];
        o[t] = __builtin_amdgcn_wmma_f32_16x16x32_bf16(false, pf.v, false,
                                                       vf.v, (short)0, o[t],
                                                       false, false);
      }
    }
    __syncthreads();
  }

#pragma unroll
  for (int t = 0; t < 4; ++t)
#pragma unroll
    for (int v = 0; v < 8; ++v) {
      int row = q0 + v + 8 * hl;
      int col = h * 64 + t * 16 + l15;
      Ob[((size_t)b * 2048 + row) * 1024 + col] = f2bf(o[t][v] / lrun[v]);
    }
}

// ---------------------------------------------------------------- launch
extern "C" void kernel_launch(void* const* d_in, const int* in_sizes, int n_in,
                              void* d_out, int out_size, void* d_ws, size_t ws_size,
                              hipStream_t stream)
{
  const float* x  = (const float*)d_in[0];
  const float* qw = (const float*)d_in[1];
  const float* kw = (const float*)d_in[2];
  const float* vw = (const float*)d_in[3];
  const float* ow = (const float*)d_in[4];

  char* ws = (char*)d_ws;
  const size_t MB = 1024ull * 1024ull;
  unsigned short* XB = (unsigned short*)(ws);             // x bf16, 16 MB
  unsigned short* WQ = (unsigned short*)(ws + 16 * MB);
  unsigned short* WK = (unsigned short*)(ws + 18 * MB);
  unsigned short* WV = (unsigned short*)(ws + 20 * MB);
  unsigned short* WO = (unsigned short*)(ws + 22 * MB);
  unsigned short* QB = (unsigned short*)(ws + 24 * MB);   // [B,H,S,64] bf16
  unsigned short* KB = (unsigned short*)(ws + 40 * MB);
  unsigned short* VB = (unsigned short*)(ws + 56 * MB);
  unsigned short* AB = (unsigned short*)(ws + 72 * MB);   // attn out bf16

  const int nx = 8192 * 1024, nw = 1024 * 1024;
  f32_to_bf16_kernel<<<nx / 256, 256, 0, stream>>>(x,  XB, nx);
  f32_to_bf16_kernel<<<nw / 256, 256, 0, stream>>>(qw, WQ, nw);
  f32_to_bf16_kernel<<<nw / 256, 256, 0, stream>>>(kw, WK, nw);
  f32_to_bf16_kernel<<<nw / 256, 256, 0, stream>>>(vw, WV, nw);
  f32_to_bf16_kernel<<<nw / 256, 256, 0, stream>>>(ow, WO, nw);

  dim3 g(64, 16), blk(256);
  wmma_gemm_kernel<0><<<g, blk, 0, stream>>>(XB, WQ, QB);
  wmma_gemm_kernel<0><<<g, blk, 0, stream>>>(XB, WK, KB);
  wmma_gemm_kernel<0><<<g, blk, 0, stream>>>(XB, WV, VB);

  rope_kernel<<<(4 * 16 * 2048 * 32) / 256, 256, 0, stream>>>(QB, KB);

  flash_attn_kernel<<<dim3(64, 16), 256, 0, stream>>>(QB, KB, VB, AB);

  wmma_gemm_kernel<1><<<g, blk, 0, stream>>>(AB, WO, (float*)d_out);
}
